// AttentionLayer_80668075753997
// MI455X (gfx1250) — matrix operand order repaired and checked by reading, not compiled, not run on hardware
//
#include <hip/hip_runtime.h>
#include <hip/hip_bf16.h>

// Problem constants (from reference setup_inputs)
#define BATCH 4
#define SEQ   2048
#define DMODEL 1024
#define NHEAD 16
#define HDIM  64   // DMODEL / NHEAD

typedef __attribute__((ext_vector_type(16))) __bf16 v16bf;
typedef __attribute__((ext_vector_type(8)))  __bf16 v8bf;
typedef __attribute__((ext_vector_type(8)))  float  v8f;
typedef __attribute__((ext_vector_type(4)))  float  v4f;

// Load 8 contiguous bf16 (16B) into half of a WMMA operand register.
__device__ __forceinline__ void ld8(v16bf& v, int base, const __bf16* __restrict__ p) {
    v8bf t = *(const v8bf*)p;
#pragma unroll
    for (int i = 0; i < 8; ++i) v[base + i] = t[i];
}

__device__ __forceinline__ v8f wmma_bf16(v16bf a, v16bf b, v8f c) {
    return __builtin_amdgcn_wmma_f32_16x16x32_bf16(
        /*neg_a=*/false, a, /*neg_b=*/false, b,
        /*c_mod=*/(short)0, c, /*reuse_a=*/false, /*reuse_b=*/false);
}

// Load + convert a 16x32 f32 A-tile slice for one lane into WMMA A layout.
__device__ __forceinline__ v16bf ldA_f32(const float* __restrict__ row, int k0, int half) {
    const int ka = k0 + half * 8;
    v4f x0 = *(const v4f*)(row + ka);
    v4f x1 = *(const v4f*)(row + ka + 4);
    v4f x2 = *(const v4f*)(row + ka + 16);
    v4f x3 = *(const v4f*)(row + ka + 20);
    v16bf a;
#pragma unroll
    for (int i = 0; i < 4; ++i) {
        a[i]      = (__bf16)x0[i];
        a[4 + i]  = (__bf16)x1[i];
        a[8 + i]  = (__bf16)x2[i];
        a[12 + i] = (__bf16)x3[i];
    }
    return a;
}

// Load + convert a 32x16 f32 B-tile slice (contiguous K from one W row).
__device__ __forceinline__ v16bf ldB_f32(const float* __restrict__ row, int k0, int half) {
    const int kb = k0 + half * 8;
    v4f w0 = *(const v4f*)(row + kb);
    v4f w1 = *(const v4f*)(row + kb + 4);
    v4f w2 = *(const v4f*)(row + kb + 16);
    v4f w3 = *(const v4f*)(row + kb + 20);
    v16bf b;
#pragma unroll
    for (int i = 0; i < 4; ++i) {
        b[i]      = (__bf16)w0[i];
        b[4 + i]  = (__bf16)w1[i];
        b[8 + i]  = (__bf16)w2[i];
        b[12 + i] = (__bf16)w3[i];
    }
    return b;
}

// Load a bf16 K/V row slice as a WMMA B operand (8 K-values at p, 8 at p+16).
__device__ __forceinline__ v16bf ldB_bf16(const __bf16* __restrict__ p) {
    v16bf b;
    ld8(b, 0, p);
    ld8(b, 8, p + 16);
    return b;
}

// ---------------------------------------------------------------------------
// Kernel 1: Y = X @ W^T + bias, 32x64 output tile per wave.
// All 6 operand tiles are loaded into distinct registers, then 8 WMMAs are
// issued back-to-back.  grid = (B*S/32, D/64), block = 32.
// ---------------------------------------------------------------------------
__global__ void qkv_proj_kernel(const float* __restrict__ X,   // [B*S, D]
                                const float* __restrict__ W,   // [D, D]
                                const float* __restrict__ bias,// [D]
                                __bf16* __restrict__ out)      // [B, H, S, hd]
{
    const int lane = threadIdx.x;
    const int half = lane >> 4;
    const int l15  = lane & 15;
    const int rt = blockIdx.x;    // 32-row tile of B*S
    const int ct = blockIdx.y;    // 64-col tile of D == head index

    const float* xrow0 = X + (size_t)(rt * 32 + l15) * DMODEL;
    const float* xrow1 = xrow0 + (size_t)16 * DMODEL;
    const float* wrow[4];
#pragma unroll
    for (int nt = 0; nt < 4; ++nt)
        wrow[nt] = W + (size_t)(ct * 64 + nt * 16 + l15) * DMODEL;

    v8f acc[8] = {{}, {}, {}, {}, {}, {}, {}, {}};

    for (int k0 = 0; k0 < DMODEL; k0 += 32) {
        // Unconditional speculative prefetch of next K-step (global_prefetch_b8,
        // near-temporal). OOB prefetch on the final step is silently dropped.
        __builtin_prefetch(xrow0 + k0 + 32, 0, 3);
        __builtin_prefetch(xrow1 + k0 + 32, 0, 3);
        // Batch all operand loads/converts into distinct registers...
        const v16bf a0 = ldA_f32(xrow0, k0, half);
        const v16bf a1 = ldA_f32(xrow1, k0, half);
        v16bf b[4];
#pragma unroll
        for (int nt = 0; nt < 4; ++nt) b[nt] = ldB_f32(wrow[nt], k0, half);
        // ...then one uninterrupted run of 8 WMMAs.
#pragma unroll
        for (int nt = 0; nt < 4; ++nt) acc[nt]     = wmma_bf16(a0, b[nt], acc[nt]);
#pragma unroll
        for (int nt = 0; nt < 4; ++nt) acc[4 + nt] = wmma_bf16(a1, b[nt], acc[4 + nt]);
    }

    const int bidx  = (rt * 32) / SEQ;     // tile never crosses batch boundary
    const int sbase = (rt * 32) % SEQ;
    __bf16* obase = out + ((size_t)(bidx * NHEAD + ct)) * SEQ * HDIM;
#pragma unroll
    for (int nt = 0; nt < 4; ++nt) {
        const float bv = bias[ct * 64 + nt * 16 + l15];
        const int off  = nt * 16 + l15;
#pragma unroll
        for (int i = 0; i < 8; ++i) {
            const int m = half * 8 + i;                 // C layout row
            obase[(size_t)(sbase + m) * HDIM + off]      = (__bf16)(acc[nt][i] + bv);
            obase[(size_t)(sbase + 16 + m) * HDIM + off] = (__bf16)(acc[4 + nt][i] + bv);
        }
    }
}

// ---------------------------------------------------------------------------
// Kernel 2: per (bh, 32-row q-tile): row max m and raw row sum r of scores.
// 32 keys per iteration; 4 B-tiles loaded, then 8 WMMAs back-to-back.
// grid = (S/32, B*H), block = 32.
// ---------------------------------------------------------------------------
__global__ void attn_stats_kernel(const __bf16* __restrict__ Qbf,
                                  const __bf16* __restrict__ Kbf,
                                  float* __restrict__ mrow,
                                  float* __restrict__ rrow)
{
    const int lane = threadIdx.x;
    const int half = lane >> 4;
    const int l15  = lane & 15;
    const int qt = blockIdx.x;    // 32 query rows
    const int bh = blockIdx.y;

    const __bf16* qbase = Qbf + (size_t)bh * SEQ * HDIM;
    const __bf16* kbase = Kbf + (size_t)bh * SEQ * HDIM;

    v16bf a[4];   // [qq*2 + khalf]: qq sub-tile, k 0..31 / 32..63
#pragma unroll
    for (int qq = 0; qq < 2; ++qq) {
        const __bf16* qrow = qbase + (size_t)(qt * 32 + qq * 16 + l15) * HDIM;
        ld8(a[qq * 2], 0, qrow + half * 8);
        ld8(a[qq * 2], 8, qrow + 16 + half * 8);
        ld8(a[qq * 2 + 1], 0, qrow + 32 + half * 8);
        ld8(a[qq * 2 + 1], 8, qrow + 48 + half * 8);
    }

    float mx[2][8], sm[2][8];
#pragma unroll
    for (int qq = 0; qq < 2; ++qq)
#pragma unroll
        for (int i = 0; i < 8; ++i) { mx[qq][i] = -3.4e38f; sm[qq][i] = 0.0f; }

    for (int p = 0; p < SEQ / 32; ++p) {
        __builtin_prefetch(kbase + (size_t)((p + 1) * 32 + l15) * HDIM, 0, 3);
        // Load B operands for both 16-key tiles of this 32-key chunk.
        v16bf b[2][2];   // [t][khalf]
#pragma unroll
        for (int t = 0; t < 2; ++t) {
            const __bf16* krow = kbase + (size_t)(p * 32 + t * 16 + l15) * HDIM + half * 8;
            b[t][0] = ldB_bf16(krow);
            b[t][1] = ldB_bf16(krow + 32);
        }
        // 8 WMMAs into 4 score accumulators.
        v8f c[2][2] = {{{}, {}}, {{}, {}}};   // [t][qq]
#pragma unroll
        for (int t = 0; t < 2; ++t)
#pragma unroll
            for (int qq = 0; qq < 2; ++qq) {
                c[t][qq] = wmma_bf16(a[qq * 2],     b[t][0], c[t][qq]);
                c[t][qq] = wmma_bf16(a[qq * 2 + 1], b[t][1], c[t][qq]);
            }
#pragma unroll
        for (int t = 0; t < 2; ++t)
#pragma unroll
            for (int qq = 0; qq < 2; ++qq)
#pragma unroll
                for (int i = 0; i < 8; ++i) {
                    const float sv = c[t][qq][i] * 0.125f;   // 1/sqrt(64)
                    mx[qq][i] = fmaxf(mx[qq][i], sv);
                    sm[qq][i] += sv;
                }
    }
#pragma unroll
    for (int off = 1; off < 16; off <<= 1) {
#pragma unroll
        for (int qq = 0; qq < 2; ++qq)
#pragma unroll
            for (int i = 0; i < 8; ++i) {
                mx[qq][i] = fmaxf(mx[qq][i], __shfl_xor(mx[qq][i], off, 32));
                sm[qq][i] += __shfl_xor(sm[qq][i], off, 32);
            }
    }
    if (l15 == 0) {
#pragma unroll
        for (int qq = 0; qq < 2; ++qq) {
            const int rbase = bh * SEQ + qt * 32 + qq * 16 + half * 8;
#pragma unroll
            for (int i = 0; i < 8; ++i) {
                mrow[rbase + i] = mx[qq][i];
                rrow[rbase + i] = sm[qq][i];
            }
        }
    }
}

// ---------------------------------------------------------------------------
// Kernel 3: O = (exp(S - m)/r) @ V; 32 query rows x 64 v-dims per wave,
// streamed over 32-key chunks. Score WMMAs batched; P transposed via LDS;
// V transposed via LDS for the B operand of the P@V WMMAs.
// grid = (S/32, B*H), block = 32.
// ---------------------------------------------------------------------------
__global__ void attn_out_kernel(const __bf16* __restrict__ Qbf,
                                const __bf16* __restrict__ Kbf,
                                const __bf16* __restrict__ Vbf,
                                const float* __restrict__ mrow,
                                const float* __restrict__ rrow,
                                float* __restrict__ out)   // [B, S, D]
{
    __shared__ __bf16 Plds[32][32];   // P tile, [q-row][key-in-chunk]
    __shared__ __bf16 Vlds[64][32];   // V transposed, [vdim][key-in-chunk]

    const int lane = threadIdx.x;
    const int half = lane >> 4;
    const int l15  = lane & 15;
    const int qt = blockIdx.x;    // 32 query rows
    const int bh = blockIdx.y;

    const __bf16* qbase = Qbf + (size_t)bh * SEQ * HDIM;
    const __bf16* kbase = Kbf + (size_t)bh * SEQ * HDIM;
    const __bf16* vbase = Vbf + (size_t)bh * SEQ * HDIM;

    v16bf a[4];
#pragma unroll
    for (int qq = 0; qq < 2; ++qq) {
        const __bf16* qrow = qbase + (size_t)(qt * 32 + qq * 16 + l15) * HDIM;
        ld8(a[qq * 2], 0, qrow + half * 8);
        ld8(a[qq * 2], 8, qrow + 16 + half * 8);
        ld8(a[qq * 2 + 1], 0, qrow + 32 + half * 8);
        ld8(a[qq * 2 + 1], 8, qrow + 48 + half * 8);
    }

    float mr[2][8], ri[2][8];
#pragma unroll
    for (int qq = 0; qq < 2; ++qq) {
        const int rbase = bh * SEQ + qt * 32 + qq * 16 + half * 8;
#pragma unroll
        for (int i = 0; i < 8; ++i) {
            mr[qq][i] = mrow[rbase + i];
            ri[qq][i] = 1.0f / rrow[rbase + i];   // faithful to buggy softmax
        }
    }

    v8f oacc[8] = {{}, {}, {}, {}, {}, {}, {}, {}};  // [qq*4 + nt]

    for (int p = 0; p < SEQ / 32; ++p) {
        __builtin_prefetch(kbase + (size_t)((p + 1) * 32 + l15) * HDIM, 0, 3);
        // --- scores: batch B loads, then 8 WMMAs, then exp -> Plds ---
        v16bf b[2][2];
#pragma unroll
        for (int t = 0; t < 2; ++t) {
            const __bf16* krow = kbase + (size_t)(p * 32 + t * 16 + l15) * HDIM + half * 8;
            b[t][0] = ldB_bf16(krow);
            b[t][1] = ldB_bf16(krow + 32);
        }
        v8f c[2][2] = {{{}, {}}, {{}, {}}};
#pragma unroll
        for (int t = 0; t < 2; ++t)
#pragma unroll
            for (int qq = 0; qq < 2; ++qq) {
                c[t][qq] = wmma_bf16(a[qq * 2],     b[t][0], c[t][qq]);
                c[t][qq] = wmma_bf16(a[qq * 2 + 1], b[t][1], c[t][qq]);
            }
#pragma unroll
        for (int t = 0; t < 2; ++t)
#pragma unroll
            for (int qq = 0; qq < 2; ++qq)
#pragma unroll
                for (int i = 0; i < 8; ++i) {
                    const float sv = c[t][qq][i] * 0.125f;
                    const float pv = __expf(sv - mr[qq][i]) * ri[qq][i];
                    Plds[qq * 16 + half * 8 + i][t * 16 + l15] = (__bf16)pv;
                }
        // --- stage V transposed: one key row per lane ---
        {
            const __bf16* vrw = vbase + (size_t)(p * 32 + lane) * HDIM;
            __builtin_prefetch(vrw + 32 * HDIM, 0, 3);
#pragma unroll
            for (int c8 = 0; c8 < 8; ++c8) {
                v8bf t = *(const v8bf*)(vrw + c8 * 8);
#pragma unroll
                for (int e = 0; e < 8; ++e) Vlds[c8 * 8 + e][lane] = t[e];
            }
        }
        __syncthreads();

        // --- P@V: batch pa/vb loads, then 8 WMMAs back-to-back ---
        v16bf pa[2], vb[4];
#pragma unroll
        for (int qq = 0; qq < 2; ++qq) {
            const __bf16* prow = &Plds[qq * 16 + l15][0];
            ld8(pa[qq], 0, prow + half * 8);
            ld8(pa[qq], 8, prow + 16 + half * 8);
        }
#pragma unroll
        for (int nt = 0; nt < 4; ++nt)
            vb[nt] = ldB_bf16(&Vlds[nt * 16 + l15][half * 8]);
#pragma unroll
        for (int qq = 0; qq < 2; ++qq)
#pragma unroll
            for (int nt = 0; nt < 4; ++nt)
                oacc[qq * 4 + nt] = wmma_bf16(pa[qq], vb[nt], oacc[qq * 4 + nt]);
        __syncthreads();
    }

    // merged-head write: out[b, s, h*64 + off]
    const int b = bh / NHEAD;
    const int h = bh % NHEAD;
#pragma unroll
    for (int qq = 0; qq < 2; ++qq) {
#pragma unroll
        for (int nt = 0; nt < 4; ++nt) {
#pragma unroll
            for (int i = 0; i < 8; ++i) {
                const int s   = qt * 32 + qq * 16 + half * 8 + i;
                const int col = h * HDIM + nt * 16 + l15;
                out[((size_t)b * SEQ + s) * DMODEL + col] = oacc[qq * 4 + nt][i];
            }
        }
    }
}

// ---------------------------------------------------------------------------
// Host launcher. Workspace: Qbf/Kbf/Vbf (16 MB each, bf16 [B,H,S,hd]) then
// mrow/rrow (0.5 MB each f32).  Total ~49 MB.
// ---------------------------------------------------------------------------
extern "C" void kernel_launch(void* const* d_in, const int* in_sizes, int n_in,
                              void* d_out, int out_size, void* d_ws, size_t ws_size,
                              hipStream_t stream) {
    (void)in_sizes; (void)n_in; (void)out_size; (void)ws_size;
    const float* X  = (const float*)d_in[0];
    const float* Wq = (const float*)d_in[1];
    const float* bq = (const float*)d_in[2];
    const float* Wk = (const float*)d_in[3];
    const float* bk = (const float*)d_in[4];
    const float* Wv = (const float*)d_in[5];
    const float* bv = (const float*)d_in[6];
    float* out = (float*)d_out;

    char* ws = (char*)d_ws;
    const size_t qkvBytes = (size_t)BATCH * NHEAD * SEQ * HDIM * sizeof(__bf16);
    __bf16* Qbf = (__bf16*)(ws);
    __bf16* Kbf = (__bf16*)(ws + qkvBytes);
    __bf16* Vbf = (__bf16*)(ws + 2 * qkvBytes);
    float*  mrow = (float*)(ws + 3 * qkvBytes);
    float*  rrow = mrow + (size_t)BATCH * NHEAD * SEQ;

    dim3 blk(32);
    dim3 g1((BATCH * SEQ) / 32, DMODEL / 64);
    qkv_proj_kernel<<<g1, blk, 0, stream>>>(X, Wq, bq, Qbf);
    qkv_proj_kernel<<<g1, blk, 0, stream>>>(X, Wk, bk, Kbf);
    qkv_proj_kernel<<<g1, blk, 0, stream>>>(X, Wv, bv, Vbf);

    dim3 g2(SEQ / 32, BATCH * NHEAD);
    attn_stats_kernel<<<g2, blk, 0, stream>>>(Qbf, Kbf, mrow, rrow);
    attn_out_kernel<<<g2, blk, 0, stream>>>(Qbf, Kbf, Vbf, mrow, rrow, out);
}
